// AngledFBP_42666205118735
// MI455X (gfx1250) — matrix operand order repaired and checked by reading, not compiled, not run on hardware
//
#include <hip/hip_runtime.h>
#include <math.h>

typedef __attribute__((ext_vector_type(16))) _Float16 v16h;
typedef __attribute__((ext_vector_type(8)))  float    v8f;

#define NS 64    // slices (z)
#define NA 180   // angles
#define DM 512   // detector width / recon width
#define CH 16    // z-chunk per backprojection thread

#define FS_BYTES ((size_t)NA * NS * DM * sizeof(float))   // 23,592,960

__device__ __forceinline__ float ramp_g(int d) {
  // spatial ramp kernel g = 2*f : g(0)=0.5, g(odd d) = -2/(pi d)^2, g(even d)=0
  if (d == 0) return 0.5f;
  if ((d & 1) == 0) return 0.0f;
  float pd = 3.14159265358979323846f * (float)d;
  return -2.0f / (pd * pd);
}

// ---------------------------------------------------------------------------
// Kernel 0: build the symmetric ramp Toeplitz matrix in f16 (512 KB, L2-res).
//   Gh[n][k] = g(|n-k|)   (symmetric, so this serves as B[k][n] row-major in k)
// ---------------------------------------------------------------------------
__global__ __launch_bounds__(256) void build_ramp(_Float16* __restrict__ Gh) {
  const int k = blockIdx.x * blockDim.x + threadIdx.x;  // 0..511
  const int n = blockIdx.y;                             // 0..511
  int d = n - k;
  if (d < 0) d = -d;
  Gh[(size_t)n * DM + k] = (_Float16)ramp_g(d);
}

// ---------------------------------------------------------------------------
// Kernel 1: ramp filtering as a WMMA GEMM.
//   Y[a][s][i] = sum_j X[s][a][j] * G[j][i]
// Rows r = a*NS + s (11520 rows), cols i (512), K = j (512).
// One wave32 per 16x16 output tile, v_wmma_f32_16x16x32_f16, 16 K-steps.
// 16 | 64, so a 16-row tile never crosses a slice-group: a = tileR>>2,
// s = (tileR&3)*16 + rowInTile  (no runtime div/mod).
// ---------------------------------------------------------------------------
__global__ __launch_bounds__(256) void fbp_filter_wmma(
    const float* __restrict__ sino,      // [NS][1][NA][DM]
    const _Float16* __restrict__ Gh,     // [DM][DM] f16
    float* __restrict__ fs)              // [NA][NS][DM]
{
  const int lane = threadIdx.x & 31;
  const int wave = threadIdx.x >> 5;
  const int tile = blockIdx.x * 8 + wave;   // 23040 tiles total
  const int tileR = tile >> 5;              // 0..719  (16-row tiles of 11520)
  const int tileC = tile & 31;              // 0..31   (16-col tiles of 512)

  const int l16 = lane & 15;
  const int hi  = lane >> 4;

  const int a     = tileR >> 2;             // angle index (0..179)
  const int sbase = (tileR & 3) * 16;       // first slice of this tile

  // A-fragment source row for this lane
  const float* __restrict__ rowp = sino + ((size_t)(sbase + l16) * NA + a) * DM;

  const int ncol = tileC * 16 + l16;        // B/D column for this lane
  // B fragment base: elements 0..15 walk K = kk + 16*hi + (0..15) contiguously
  const _Float16* __restrict__ brow = Gh + (size_t)ncol * DM + 16 * hi;

  v8f acc = {};

  for (int kk = 0; kk < DM; kk += 32) {
    v16h afrag;

    // A fragment (16-bit A 16x32 layout): pairs of consecutive K -> float2 load
#pragma unroll
    for (int v = 0; v < 8; ++v) {
      int k0 = kk + 16 * hi + 2 * v;
      float2 x = *(const float2*)(rowp + k0);
      afrag[2 * v]     = (_Float16)x.x;
      afrag[2 * v + 1] = (_Float16)x.y;
    }

    // B fragment (32x16 layout): 16 contiguous halfs (two b128 loads)
    v16h bfrag = *(const v16h*)(brow + kk);

    acc = __builtin_amdgcn_wmma_f32_16x16x32_f16(
        /*neg_a=*/false, afrag, /*neg_b=*/false, bfrag,
        /*c_mod=*/(short)0, acc, /*reuse_a=*/false, /*reuse_b=*/false);
  }

  // D store: VGPR v holds row M = v + 8*hi -> slice sbase + v + 8*hi, col ncol
  float* __restrict__ outp =
      fs + ((size_t)a * NS + sbase + 8 * hi) * DM + ncol;
#pragma unroll
  for (int v = 0; v < 8; ++v) {
    outp[(size_t)v * DM] = acc[v];
  }
}

// ---------------------------------------------------------------------------
// Kernel 2: tilted backprojection, one thread per 16-deep z-column.
// zpix(z) = z + 31.5*zd exactly (linspace step * 31.5 == 1), so wz is
// constant along z and sampled rows slide by one: 17 row-samples (2 gathers
// each) cover 16 voxels -> 2.125 gathers/voxel instead of 4, and all
// trig/address work per angle is amortized over the column.
// ---------------------------------------------------------------------------
__global__ __launch_bounds__(256) void fbp_backproject(
    const float* __restrict__ fs,     // [NA][NS][DM]
    const float* __restrict__ theta,  // [NA]
    const float* __restrict__ tiltp,  // [1]
    float* __restrict__ out)          // [NS][DM][DM]
{
  __shared__ float s_cos[NA];
  __shared__ float s_sin[NA];
  for (int i = threadIdx.x; i < NA; i += blockDim.x) {
    float t = theta[i];
    float sv, cv;
    __sincosf(t, &sv, &cv);
    s_cos[i] = cv;
    s_sin[i] = sv;
  }
  __syncthreads();

  const int x     = blockIdx.x * blockDim.x + threadIdx.x;  // 0..511 (col)
  const int y     = blockIdx.y;                             // 0..511 (row)
  const int zbase = blockIdx.z * CH;                        // 0,16,32,48

  const float gxv = -1.0f + 2.0f * (float)x / 511.0f;
  const float gyv = -1.0f + 2.0f * (float)y / 511.0f;
  const float st  = __sinf(tiltp[0]);
  const float mn  = (float)DM / (float)NS;                  // 8.0

  float acc[CH];
#pragma unroll
  for (int u = 0; u < CH; ++u) acc[u] = 0.0f;

  for (int aa = 0; aa < NA; ++aa) {
    const float c  = s_cos[aa];
    const float sn = s_sin[aa];
    const float rx = gxv * c - gyv * sn;
    const float zd = (gxv * sn + gyv * c) * st * mn;

    // detector coordinate (shared by the whole z-column)
    const float xp  = (rx + 1.0f) * 0.5f * (float)(DM - 1);
    const float x0f = floorf(xp);
    const float wx  = xp - x0f;
    const int x0 = (int)x0f;
    const int x1 = x0 + 1;
    const float fx0 = (x0 >= 0 && x0 < DM) ? 1.0f : 0.0f;
    const float fx1 = (x1 >= 0 && x1 < DM) ? 1.0f : 0.0f;
    const int xc0 = min(max(x0, 0), DM - 1);
    const int xc1 = min(max(x1, 0), DM - 1);
    const float w0 = fx0 * (1.0f - wx);   // weight of column x0
    const float w1 = fx1 * wx;            // weight of column x1

    // z coordinate: zp(z) = z + t, t = 31.5*zd ; constant fraction wz
    const float t   = zd * (0.5f * (float)(NS - 1));
    const float t0f = floorf(t);
    const float wz  = t - t0f;
    const float onemwz = 1.0f - wz;
    const int rbase = zbase + (int)t0f;

    const float* __restrict__ img = fs + (size_t)aa * NS * DM;

#pragma unroll
    for (int q = 0; q <= CH; ++q) {
      const int zr = rbase + q;
      const int zc = min(max(zr, 0), NS - 1);
      float rowv = img[zc * DM + xc0] * w0 + img[zc * DM + xc1] * w1;
      rowv = (zr >= 0 && zr < NS) ? rowv : 0.0f;
      if (q > 0)  acc[q - 1] += rowv * wz;
      if (q < CH) acc[q]     += rowv * onemwz;
    }
  }

  const float r2 = gxv * gxv + gyv * gyv;
  const float scale = (r2 > 0.95f)
                        ? 0.0f
                        : 3.14159265358979323846f / (2.0f * (float)NA);
  float* __restrict__ op = out + ((size_t)zbase * DM + y) * DM + x;
#pragma unroll
  for (int u = 0; u < CH; ++u) {
    op[(size_t)u * DM * DM] = acc[u] * scale;
  }
}

extern "C" void kernel_launch(void* const* d_in, const int* in_sizes, int n_in,
                              void* d_out, int out_size, void* d_ws, size_t ws_size,
                              hipStream_t stream) {
  const float* sino  = (const float*)d_in[0];  // [64][1][180][512]
  const float* tilt  = (const float*)d_in[1];  // [1]
  const float* theta = (const float*)d_in[2];  // [180]
  float* out = (float*)d_out;                  // [64][512][512]

  float*     fs = (float*)d_ws;                            // [180][64][512] filtered
  _Float16*  Gh = (_Float16*)((char*)d_ws + FS_BYTES);     // [512][512] f16 ramp matrix

  // 0) build 512x512 f16 ramp Toeplitz matrix (once, 512 KB)
  build_ramp<<<dim3(DM / 256, DM), dim3(256), 0, stream>>>(Gh);

  // 1) WMMA filter GEMM: 23040 16x16 tiles, 8 waves (tiles) per 256-thread block
  fbp_filter_wmma<<<dim3(2880), dim3(256), 0, stream>>>(sino, Gh, fs);

  // 2) backprojection: one thread per 16-deep z-column, grid (512/256, 512, 4)
  fbp_backproject<<<dim3(DM / 256, DM, NS / CH), dim3(256), 0, stream>>>(fs, theta, tilt, out);
}
